// Decorrelation_69922067579089
// MI455X (gfx1250) — compile-verified, matrix operand order checked by reading, not run on hardware
//
#include <hip/hip_runtime.h>

typedef __attribute__((ext_vector_type(16))) _Float16 v16h;
typedef __attribute__((ext_vector_type(8)))  _Float16 v8h;
typedef __attribute__((ext_vector_type(8)))  float    v8f;

#define NV     32
#define KDIM   512      // padded K: 31 covariates * 16 coeffs, + 16 zeros
#define NPAIR  496

// ---------------------------------------------------------------------------
// Kernel 1: pack params (16 x 496, f32) into f16 B matrix, column-major [v][k]
// k = c*16 + i ; value = params[i, pair(v,c)] if c < v else 0
// ---------------------------------------------------------------------------
__global__ void build_B_kernel(const float* __restrict__ params,
                               _Float16* __restrict__ Bg) {
  int e = blockIdx.x * blockDim.x + threadIdx.x;   // 0 .. 32*512-1
  int v = e >> 9;        // output column 0..31
  int k = e & 511;       // padded K index
  int c = k >> 4;        // covariate 0..31
  int i = k & 15;        // coefficient row 0..15
  float val = 0.0f;
  if (c < v) {           // (c==31 impossible since v<=31)
    int p = (v * (v - 1)) / 2 + c;
    val = params[i * NPAIR + p];
  }
  Bg[e] = (_Float16)val;
}

// ---------------------------------------------------------------------------
// Kernel 2: fused basis construction + WMMA GEMM + epilogue.
// Block = 64 threads (2 waves). Each wave owns a 16-row tile.
// B staged to LDS with async global->LDS copies overlapped with A-tile build.
// ---------------------------------------------------------------------------
__global__ __launch_bounds__(64)
void decorr_wmma_kernel(const float* __restrict__ x,
                        const _Float16* __restrict__ Bg,
                        float* __restrict__ out) {
  __shared__ _Float16 Bl[NV * KDIM];        // 32 KB  [v][k]
  __shared__ _Float16 Al[2][16 * KDIM];     // 2 x 16 KB, row-major [row][k]

  const int tid  = threadIdx.x;
  const int wave = tid >> 5;
  const int lane = tid & 31;
  const int rowBase = blockIdx.x * 32 + wave * 16;

  // ---- issue async B copy: 32768 B = 2048 x b128, ASYNCcnt-tracked,
  //      no VGPR round-trip; overlapped with the A-tile build below ----
  {
    unsigned ldsBase = (unsigned)(size_t)&Bl[0];
    for (int i = tid; i < 2048; i += 64) {
      unsigned lds  = ldsBase + i * 16;   // LDS byte address (VDST)
      unsigned voff = i * 16;             // byte offset from saddr (VADDR)
      asm volatile("global_load_async_to_lds_b128 %0, %1, %2 offset:0"
                   :: "v"(lds), "v"(voff), "s"(Bg)
                   : "memory");
    }
  }

  // ---- zero this wave's A tile (16 KB = 1024 uint4, 32 lanes) ----
  {
    uint4* dst = (uint4*)Al[wave];
    uint4  z{0, 0, 0, 0};
    for (int i = lane; i < 1024; i += 32) dst[i] = z;
  }

  // ---- scatter basis*x into A tile: tasks t -> (c = t>>4 in 0..30, row = t&15)
  // Uniform cubic B-spline: s = (x-t0)/d - k, k = clamp(floor((x-t0)/d),3,15)
  const float t0   = -19.0f / 3.0f;
  const float invd = 1.5f;
  for (int t = lane; t < 496; t += 32) {
    int   c   = t >> 4;
    int   row = t & 15;
    float xv  = x[(rowBase + row) * NV + c];
    float u   = (xv - t0) * invd;
    float fk  = floorf(u);
    fk        = fminf(fmaxf(fk, 3.0f), 15.0f);
    int   k   = (int)fk;
    float s   = u - fk;
    float s2 = s * s, s3 = s2 * s;
    const float sixth = 1.0f / 6.0f;
    float w0 = sixth * (-s3 + 3.0f * s2 - 3.0f * s + 1.0f);
    float w1 = sixth * (3.0f * s3 - 6.0f * s2 + 4.0f);
    float w2 = sixth * (-3.0f * s3 + 3.0f * s2 + 3.0f * s + 1.0f);
    float w3 = sixth * s3;
    _Float16* dst = &Al[wave][row * KDIM + c * 16 + (k - 3)];
    dst[0] = (_Float16)(w0 * xv);
    dst[1] = (_Float16)(w1 * xv);
    dst[2] = (_Float16)(w2 * xv);
    dst[3] = (_Float16)(w3 * xv);
  }

  // B async copy must be complete & visible to all waves before GEMM
  asm volatile("s_wait_asynccnt 0" ::: "memory");
  __syncthreads();

  // ---- GEMM: [16 x 512] x [512 x 32] via v_wmma_f32_16x16x32_f16 ----
  // Fragment layout (16-bit, wave32): lanes 0-15 hold K {0..7,16..23},
  // lanes 16-31 hold K {8..15,24..31}; A row / B col = lane & 15.
  v8f acc0 = {};
  v8f acc1 = {};
  const int m   = lane & 15;
  const int off = (lane >> 4) * 8;
  for (int kk = 0; kk < 16; ++kk) {
    int kb = kk * 32 + off;
    union { v16h v; v8h h[2]; } a, b0, b1;
    a.h[0]  = *(const v8h*)&Al[wave][m * KDIM + kb];
    a.h[1]  = *(const v8h*)&Al[wave][m * KDIM + kb + 16];
    b0.h[0] = *(const v8h*)&Bl[m * KDIM + kb];
    b0.h[1] = *(const v8h*)&Bl[m * KDIM + kb + 16];
    b1.h[0] = *(const v8h*)&Bl[(16 + m) * KDIM + kb];
    b1.h[1] = *(const v8h*)&Bl[(16 + m) * KDIM + kb + 16];
    acc0 = __builtin_amdgcn_wmma_f32_16x16x32_f16(false, a.v, false, b0.v,
                                                  (short)0, acc0, false, false);
    acc1 = __builtin_amdgcn_wmma_f32_16x16x32_f16(false, a.v, false, b1.v,
                                                  (short)0, acc1, false, false);
  }

  // ---- epilogue: D layout -> VGPR r holds M = r + 8*(lane>=16), N = lane&15
  const int mrow = rowBase + (lane >> 4) * 8;
  const int col  = lane & 15;
#pragma unroll
  for (int r = 0; r < 8; ++r) {
    int row = mrow + r;
    out[row * NV + col]      = x[row * NV + col]      + acc0[r];
    out[row * NV + col + 16] = x[row * NV + col + 16] + acc1[r];
  }
}

// ---------------------------------------------------------------------------
extern "C" void kernel_launch(void* const* d_in, const int* in_sizes, int n_in,
                              void* d_out, int out_size, void* d_ws, size_t ws_size,
                              hipStream_t stream) {
  const float* x      = (const float*)d_in[0];   // [N, 32] f32
  const float* params = (const float*)d_in[1];   // [16, 496] f32
  float*       out    = (float*)d_out;           // [N, 32] f32
  _Float16*    Bg     = (_Float16*)d_ws;         // 32 KB packed B matrix

  // Pack B: 32*512 = 16384 elements
  build_B_kernel<<<64, 256, 0, stream>>>(params, Bg);

  int nrows  = in_sizes[0] / NV;      // 65536
  int blocks = nrows / 32;            // 32 rows per block (2 waves x 16)
  decorr_wmma_kernel<<<blocks, 64, 0, stream>>>(x, Bg, out);
}
